// DGCNN_77841987272904
// MI455X (gfx1250) — matrix operand order repaired and checked by reading, not compiled, not run on hardware
//
#include <hip/hip_runtime.h>
#include <hip/hip_bf16.h>

#define KK    20
#define NB    2
#define NP    4096
#define BN_   (NB * NP)
#define BNK_  (BN_ * KK)
#define R3_   32768
#define SLOPE_ 0.2f

typedef __bf16 bf16;
typedef __attribute__((ext_vector_type(16))) __bf16 v16bf;
typedef __attribute__((ext_vector_type(8)))  __bf16 v8bf;
typedef __attribute__((ext_vector_type(8)))  float  v8f;

__device__ __forceinline__ float lrelu(float v) { return v > 0.f ? v : SLOPE_ * v; }

// ---------------------------------------------------------------------------
// WMMA GEMM:  Y(M,O) f32 = W(O,Kc) bf16  x  X(M,Kc [ldx]) bf16^T  (+ bias[b][o])
// Each wave computes a 64(O) x 16(M) strip = 4 WMMA tiles sharing one
// activation (B) fragment per K-step -> 4 independent accumulator chains,
// ~2x less fragment traffic per WMMA than one-tile-per-wave.
// Fragment addressing follows the CDNA5 wave32 layouts (CDNA5 ISA):
//   A 16-bit: lane(l16) = row; K runs {8*half..+7} and {16+8*half..+7}
//   B 16-bit: lane(l16) = col; K runs {8*half..+7} and {16+8*half..+7}
//   D f32:    lane(l16) = col; VGPR j -> row 8*half+j
// ---------------------------------------------------------------------------
__global__ void gemm_wmma_bf16(const bf16* __restrict__ W, const bf16* __restrict__ X,
                               float* __restrict__ Y,
                               int O, int M, int Kc, int ldx,
                               const float* __restrict__ bias, int colsPerBatch)
{
    const int wave    = blockIdx.x * (blockDim.x >> 5) + (threadIdx.x >> 5);
    const int ostrips = O >> 6;                       // strips of 64 outputs
    const int total   = ostrips * (M >> 4);
    if (wave >= total) return;                        // wave-uniform: EXEC all-ones
    const int ot   = (wave % ostrips) << 6;
    const int mt   = (wave / ostrips) << 4;
    const int lane = threadIdx.x & 31;
    const int half = lane >> 4;
    const int l16  = lane & 15;

    const bf16* wrow = W + (size_t)(ot + l16) * Kc;   // A tile 0; tiles 1..3 at +16*Kc
    const bf16* xrow = X + (size_t)(mt + l16) * ldx;

    v8f acc0 = {}, acc1 = {}, acc2 = {}, acc3 = {};
    for (int k0 = 0; k0 < Kc; k0 += 32) {
        v8bf b0 = *(const v8bf*)(xrow + k0 + 8 * half);
        v8bf b1 = *(const v8bf*)(xrow + k0 + 16 + 8 * half);
        v16bf b = __builtin_shufflevector(b0, b1, 0,1,2,3,4,5,6,7,8,9,10,11,12,13,14,15);
        #pragma unroll
        for (int t = 0; t < 4; ++t) {
            const bf16* wr = wrow + (size_t)(t * 16) * Kc;
            v8bf a0 = *(const v8bf*)(wr + k0 + 8 * half);
            v8bf a1 = *(const v8bf*)(wr + k0 + 16 + 8 * half);
            v16bf a = __builtin_shufflevector(a0, a1, 0,1,2,3,4,5,6,7,8,9,10,11,12,13,14,15);
            v8f& acc = (t == 0) ? acc0 : (t == 1) ? acc1 : (t == 2) ? acc2 : acc3;
            acc = __builtin_amdgcn_wmma_f32_16x16x32_bf16(false, a, false, b,
                                                          (short)0, acc, false, false);
        }
    }

    float* yrow = Y + (size_t)(mt + l16) * O + ot + 8 * half;
    if (bias) {
        const float* brow = bias + (size_t)((mt + l16) / colsPerBatch) * O + ot + 8 * half;
        #pragma unroll
        for (int t = 0; t < 4; ++t) {
            const v8f& acc = (t == 0) ? acc0 : (t == 1) ? acc1 : (t == 2) ? acc2 : acc3;
            #pragma unroll
            for (int j = 0; j < 8; ++j) yrow[t * 16 + j] = acc[j] + brow[t * 16 + j];
        }
    } else {
        #pragma unroll
        for (int t = 0; t < 4; ++t) {
            const v8f& acc = (t == 0) ? acc0 : (t == 1) ? acc1 : (t == 2) ? acc2 : acc3;
            #pragma unroll
            for (int j = 0; j < 8; ++j) yrow[t * 16 + j] = acc[j];
        }
    }
}

// ---------------------------------------------------------------------------
// KNN: per-thread query point, LDS tile of 128 candidates, stable top-20.
// ---------------------------------------------------------------------------
template <int C>
__global__ void knn_kernel(const float* __restrict__ X, int ldx, int* __restrict__ idx)
{
    __shared__ float tile[128][C];
    const int b = blockIdx.y;
    const int n = blockIdx.x * 128 + threadIdx.x;

    float xi[C];
    const float* xr = X + ((size_t)b * NP + n) * ldx;
    #pragma unroll
    for (int c = 0; c < C; ++c) xi[c] = xr[c];

    float best[KK]; int besti[KK];
    #pragma unroll
    for (int p = 0; p < KK; ++p) { best[p] = -3.4e38f; besti[p] = 0; }

    for (int j0 = 0; j0 < NP; j0 += 128) {
        __syncthreads();
        const float* tr = X + ((size_t)b * NP + j0 + threadIdx.x) * ldx;
        #pragma unroll
        for (int c = 0; c < C; ++c) tile[threadIdx.x][c] = tr[c];
        __syncthreads();
        for (int jj = 0; jj < 128; ++jj) {
            float d2 = 0.f;
            #pragma unroll
            for (int c = 0; c < C; ++c) { float d = xi[c] - tile[jj][c]; d2 += d * d; }
            float v = -d2; int vi = j0 + jj;
            if (v > best[KK - 1]) {
                #pragma unroll
                for (int p = 0; p < KK; ++p) {
                    if (v > best[p]) {
                        float tv = best[p]; best[p] = v;  v  = tv;
                        int   ti = besti[p]; besti[p] = vi; vi = ti;
                    }
                }
            }
        }
    }
    int* out = idx + ((size_t)b * NP + n) * KK;
    #pragma unroll
    for (int p = 0; p < KK; ++p) out[p] = besti[p];
}

// Ye[(b,n,k)][o] = Yt[b*N+idx][o] + Yt[b*N+n][64+o]   (float4 granularity)
__global__ void edge_combine(const float4* __restrict__ Yt, const int* __restrict__ idx,
                             float4* __restrict__ Ye)
{
    size_t t = (size_t)blockIdx.x * blockDim.x + threadIdx.x;
    if (t >= (size_t)BNK_ * 16) return;
    int col = (int)(t >> 4), q = (int)(t & 15);
    int b  = col / (NP * KK);
    int nk = col - b * NP * KK;
    int n  = nk / KK;
    int j  = idx[col];
    float4 a = Yt[((size_t)b * NP + j) * 32 + q];
    float4 c = Yt[((size_t)b * NP + n) * 32 + 16 + q];
    float4 r; r.x = a.x + c.x; r.y = a.y + c.y; r.z = a.z + c.z; r.w = a.w + c.w;
    Ye[(size_t)col * 16 + q] = r;
}

// per-channel sum / sumsq (coalesced; 256 chunks, atomic merge)
__global__ void bn_stats(const float* __restrict__ Y, int M, int O, float* __restrict__ sums)
{
    int t = blockIdx.x * blockDim.x + threadIdx.x;
    if (t >= O * 256) return;
    int o = t % O, chunk = t / O;
    float s = 0.f, s2 = 0.f;
    for (int m = chunk; m < M; m += 256) { float v = Y[(size_t)m * O + o]; s += v; s2 += v * v; }
    atomicAdd(&sums[o], s);
    atomicAdd(&sums[O + o], s2);
}

__global__ void bn_finalize(const float* __restrict__ sums, const float* __restrict__ gamma,
                            const float* __restrict__ beta, float* __restrict__ ss,
                            int O, float invCount)
{
    int o = blockIdx.x * blockDim.x + threadIdx.x;
    if (o >= O) return;
    float mu  = sums[o] * invCount;
    float var = sums[O + o] * invCount - mu * mu;
    float s   = gamma[o] * rsqrtf(var + 1e-5f);
    ss[o] = s; ss[O + o] = beta[o] - mu * s;
}

__global__ void bn_apply_bf16k(const float* __restrict__ Y, const float* __restrict__ ss,
                               bf16* __restrict__ Xo, int M, int O)
{
    size_t t = (size_t)blockIdx.x * blockDim.x + threadIdx.x;
    if (t >= (size_t)M * O) return;
    int o = (int)(t % O);
    Xo[t] = (bf16)lrelu(Y[t] * ss[o] + ss[O + o]);
}

// BN + LeakyReLU + max over K; writes fcat (f32, for KNN) and xcat (bf16, for GEMMs)
__global__ void bn_lrelu_maxk(const float* __restrict__ Y, const float* __restrict__ ss,
                              float* __restrict__ fcat, bf16* __restrict__ xcat, int off)
{
    int t = blockIdx.x * blockDim.x + threadIdx.x;
    if (t >= BN_ * 64) return;
    int col = t >> 6, o = t & 63;
    float s = ss[o], sh = ss[64 + o], m = -3.4e38f;
    for (int k = 0; k < KK; ++k)
        m = fmaxf(m, lrelu(Y[((size_t)col * KK + k) * 64 + o] * s + sh));
    fcat[(size_t)col * 192 + off + o] = m;
    xcat[(size_t)col * 192 + off + o] = (bf16)m;
}

// g[b][e] = max_n lrelu(bn(Y5))
__global__ void gmax_kernel(const float* __restrict__ Y5, const float* __restrict__ ss,
                            float* __restrict__ g)
{
    int t = blockIdx.x * blockDim.x + threadIdx.x;
    if (t >= NB * 1024) return;
    int b = t >> 10, e = t & 1023;
    float s = ss[e], sh = ss[1024 + e], m = -3.4e38f;
    for (int n = 0; n < NP; ++n)
        m = fmaxf(m, lrelu(Y5[((size_t)b * NP + n) * 1024 + e] * s + sh));
    g[t] = m;
}

// C6[b][o] = w6[o, :1024] . g[b]
__global__ void gcontrib_kernel(const float* __restrict__ w6, const float* __restrict__ g,
                                float* __restrict__ C6)
{
    int t = blockIdx.x * blockDim.x + threadIdx.x;
    if (t >= NB * 512) return;
    int b = t >> 9, o = t & 511;
    const float* wr = w6 + (size_t)o * 1216;
    const float* gr = g + b * 1024;
    float s = 0.f;
    for (int e = 0; e < 1024; ++e) s += wr[e] * gr[e];
    C6[t] = s;
}

// one block per point: BN+lrelu on 128 features, scatter-add into voxel grid
__global__ void voxel_scatter(const float* __restrict__ pts, const float* __restrict__ Y7,
                              const float* __restrict__ ss, float* __restrict__ acc,
                              float* __restrict__ cnt)
{
    int bn = blockIdx.x, o = threadIdx.x;
    int b = bn / NP;
    const float* p = pts + (size_t)bn * 3;
    const float invs = 1.0f / 1.101f;
    int c[3];
    #pragma unroll
    for (int d = 0; d < 3; ++d) {
        float q = p[d] * invs + 0.5f;
        q = fminf(fmaxf(q, 0.f), 0.999f);
        c[d] = (int)(q * 32.f);
    }
    int cell = c[0] + 32 * (c[1] + 32 * c[2]);
    float v = lrelu(Y7[(size_t)bn * 128 + o] * ss[o] + ss[128 + o]);
    atomicAdd(&acc[((size_t)b * R3_ + cell) * 128 + o], v);
    if (o == 0) atomicAdd(&cnt[b * R3_ + cell], 1.0f);
}

__global__ void voxel_final(const float* __restrict__ acc, const float* __restrict__ cnt,
                            float* __restrict__ out)
{
    int t = blockIdx.x * blockDim.x + threadIdx.x;
    if (t >= NB * 128 * R3_) return;
    int b = t / (128 * R3_);
    int r = t - b * 128 * R3_;
    int o = r / R3_;
    int cell = r - o * R3_;
    out[t] = acc[((size_t)b * R3_ + cell) * 128 + o] / fmaxf(cnt[b * R3_ + cell], 1.0f);
}

// weight prep: edge-conv decomposition  out = [W[:, :C] ; W[:, C:]-W[:, :C]]  zero-padded to Cpad
__global__ void prep_combine_w(const float* __restrict__ w, bf16* __restrict__ out,
                               int O, int C, int Cpad)
{
    int t = blockIdx.x * blockDim.x + threadIdx.x;
    if (t >= 2 * O * Cpad) return;
    int r = t / Cpad, c = t % Cpad;
    float v = 0.f;
    if (c < C) v = (r < O) ? w[(size_t)r * 2 * C + c]
                           : w[(size_t)(r - O) * 2 * C + C + c] - w[(size_t)(r - O) * 2 * C + c];
    out[t] = (bf16)v;
}

__global__ void prep_w_slice(const float* __restrict__ w, bf16* __restrict__ out,
                             int O, int C, int srcld, int coff)
{
    int t = blockIdx.x * blockDim.x + threadIdx.x;
    if (t >= O * C) return;
    int r = t / C, c = t % C;
    out[t] = (bf16)w[(size_t)r * srcld + coff + c];
}

__global__ void pack_points(const float* __restrict__ pts, bf16* __restrict__ out)
{
    int t = blockIdx.x * blockDim.x + threadIdx.x;
    if (t >= BN_ * 32) return;
    int r = t >> 5, c = t & 31;
    out[t] = (bf16)(c < 3 ? pts[(size_t)r * 3 + c] : 0.f);
}

// ---------------------------------------------------------------------------
extern "C" void kernel_launch(void* const* d_in, const int* in_sizes, int n_in,
                              void* d_out, int out_size, void* d_ws, size_t ws_size,
                              hipStream_t stream)
{
    (void)in_sizes; (void)n_in; (void)out_size; (void)ws_size;
    const float* pts = (const float*)d_in[0];
    const float* W[8], *G[8], *Bb[8];
    for (int i = 0; i < 8; ++i) { W[i] = (const float*)d_in[1 + 3 * i];
                                  G[i] = (const float*)d_in[2 + 3 * i];
                                  Bb[i] = (const float*)d_in[3 + 3 * i]; }

    char* ws = (char*)d_ws;
    size_t off = 0;
    auto alloc = [&](size_t bytes) { size_t o = off; off += (bytes + 255) & ~(size_t)255; return o; };
    size_t o_idx  = alloc((size_t)BNK_ * 4);
    size_t o_Yt   = alloc((size_t)BN_ * 128 * 4);
    size_t o_Ye   = alloc((size_t)BNK_ * 64 * 4);      // aliased later as voxel acc (33.5MB<=42MB)
    size_t o_Xe   = alloc((size_t)BNK_ * 64 * 2);      // aliased later as X6 bf16 (8.4MB)
    size_t o_Y2   = alloc((size_t)BNK_ * 64 * 4);      // aliased later as Y5 (33.5MB<=42MB)
    size_t o_fcat = alloc((size_t)BN_ * 192 * 4);
    size_t o_xcat = alloc((size_t)BN_ * 192 * 2);
    size_t o_Y6   = alloc((size_t)BN_ * 512 * 4);
    size_t o_Y7   = alloc((size_t)BN_ * 128 * 4);
    size_t o_cnt  = alloc((size_t)NB * R3_ * 4);
    size_t o_g    = alloc((size_t)NB * 1024 * 4);
    size_t o_C6   = alloc((size_t)NB * 512 * 4);
    size_t o_st   = alloc(2 * 1024 * 4);
    size_t o_ss   = alloc(2 * 1024 * 4);
    size_t o_Xp   = alloc((size_t)BN_ * 32 * 2);
    size_t o_Wc0  = alloc(128 * 32 * 2);
    size_t o_W1   = alloc(64 * 64 * 2);
    size_t o_Wc2  = alloc(128 * 64 * 2);
    size_t o_W3   = alloc(64 * 64 * 2);
    size_t o_Wc4  = alloc(128 * 64 * 2);
    size_t o_W5   = alloc(1024 * 192 * 2);
    size_t o_W6f  = alloc(512 * 192 * 2);
    size_t o_W7   = alloc(128 * 512 * 2);

    int*   idx  = (int*)(ws + o_idx);
    float* Yt   = (float*)(ws + o_Yt);
    float* Ye   = (float*)(ws + o_Ye);
    bf16*  Xe   = (bf16*)(ws + o_Xe);
    float* Y2   = (float*)(ws + o_Y2);
    float* fcat = (float*)(ws + o_fcat);
    bf16*  xcat = (bf16*)(ws + o_xcat);
    float* Y5   = Y2;                       // alias
    float* Y6   = (float*)(ws + o_Y6);
    float* Y7   = (float*)(ws + o_Y7);
    float* acc  = Ye;                       // alias
    float* cnt  = (float*)(ws + o_cnt);
    float* gbuf = (float*)(ws + o_g);
    float* C6   = (float*)(ws + o_C6);
    float* st   = (float*)(ws + o_st);
    float* ss   = (float*)(ws + o_ss);
    bf16*  Xp   = (bf16*)(ws + o_Xp);
    bf16*  X6   = Xe;                       // alias
    float* out  = (float*)d_out;

    const int T = 256;
    auto blk = [&](size_t n) { return (unsigned)((n + T - 1) / T); };
    auto gemm = [&](const bf16* Wb, const bf16* Xb, float* Yb, int O, int M, int Kc,
                    int ldx, const float* bias) {
        int tiles = (O >> 6) * (M >> 4);           // 64-wide output strips per wave
        gemm_wmma_bf16<<<(tiles + 3) / 4, 128, 0, stream>>>(Wb, Xb, Yb, O, M, Kc, ldx, bias, NP);
    };
    auto stats = [&](const float* Yb, int M, int O, const float* gm, const float* bt) {
        hipMemsetAsync(st, 0, 2 * O * 4, stream);
        bn_stats<<<blk((size_t)O * 256), T, 0, stream>>>(Yb, M, O, st);
        bn_finalize<<<blk(O), T, 0, stream>>>(st, gm, bt, ss, O, 1.0f / (float)M);
    };

    // ---- weight prep (bf16) ----
    prep_combine_w<<<blk(128 * 32), T, 0, stream>>>(W[0], (bf16*)(ws + o_Wc0), 64, 3, 32);
    prep_w_slice<<<blk(64 * 64), T, 0, stream>>>(W[1], (bf16*)(ws + o_W1), 64, 64, 64, 0);
    prep_combine_w<<<blk(128 * 64), T, 0, stream>>>(W[2], (bf16*)(ws + o_Wc2), 64, 64, 64);
    prep_w_slice<<<blk(64 * 64), T, 0, stream>>>(W[3], (bf16*)(ws + o_W3), 64, 64, 64, 0);
    prep_combine_w<<<blk(128 * 64), T, 0, stream>>>(W[4], (bf16*)(ws + o_Wc4), 64, 64, 64);
    prep_w_slice<<<blk(1024 * 192), T, 0, stream>>>(W[5], (bf16*)(ws + o_W5), 1024, 192, 192, 0);
    prep_w_slice<<<blk(512 * 192), T, 0, stream>>>(W[6], (bf16*)(ws + o_W6f), 512, 192, 1216, 1024);
    prep_w_slice<<<blk(128 * 512), T, 0, stream>>>(W[7], (bf16*)(ws + o_W7), 128, 512, 512, 0);
    pack_points<<<blk((size_t)BN_ * 32), T, 0, stream>>>(pts, Xp);

    dim3 kgrid(NP / 128, NB);
    size_t ecN = (size_t)BNK_ * 16;

    // ---- stage 1 ----
    knn_kernel<3><<<kgrid, 128, 0, stream>>>(pts, 3, idx);
    gemm((bf16*)(ws + o_Wc0), Xp, Yt, 128, BN_, 32, 32, nullptr);
    edge_combine<<<blk(ecN), T, 0, stream>>>((const float4*)Yt, idx, (float4*)Ye);
    stats(Ye, BNK_, 64, G[0], Bb[0]);
    bn_apply_bf16k<<<blk((size_t)BNK_ * 64), T, 0, stream>>>(Ye, ss, Xe, BNK_, 64);
    gemm((bf16*)(ws + o_W1), Xe, Y2, 64, BNK_, 64, 64, nullptr);
    stats(Y2, BNK_, 64, G[1], Bb[1]);
    bn_lrelu_maxk<<<blk((size_t)BN_ * 64), T, 0, stream>>>(Y2, ss, fcat, xcat, 0);

    // ---- stage 2 ----
    knn_kernel<64><<<kgrid, 128, 0, stream>>>(fcat, 192, idx);
    gemm((bf16*)(ws + o_Wc2), xcat, Yt, 128, BN_, 64, 192, nullptr);
    edge_combine<<<blk(ecN), T, 0, stream>>>((const float4*)Yt, idx, (float4*)Ye);
    stats(Ye, BNK_, 64, G[2], Bb[2]);
    bn_apply_bf16k<<<blk((size_t)BNK_ * 64), T, 0, stream>>>(Ye, ss, Xe, BNK_, 64);
    gemm((bf16*)(ws + o_W3), Xe, Y2, 64, BNK_, 64, 64, nullptr);
    stats(Y2, BNK_, 64, G[3], Bb[3]);
    bn_lrelu_maxk<<<blk((size_t)BN_ * 64), T, 0, stream>>>(Y2, ss, fcat, xcat, 64);

    // ---- stage 3 (single conv) ----
    knn_kernel<64><<<kgrid, 128, 0, stream>>>(fcat + 64, 192, idx);
    gemm((bf16*)(ws + o_Wc4), xcat + 64, Yt, 128, BN_, 64, 192, nullptr);
    edge_combine<<<blk(ecN), T, 0, stream>>>((const float4*)Yt, idx, (float4*)Ye);
    stats(Ye, BNK_, 64, G[4], Bb[4]);
    bn_lrelu_maxk<<<blk((size_t)BN_ * 64), T, 0, stream>>>(Ye, ss, fcat, xcat, 128);

    // ---- 1D head ----
    gemm((bf16*)(ws + o_W5), xcat, Y5, 1024, BN_, 192, 192, nullptr);
    stats(Y5, BN_, 1024, G[5], Bb[5]);
    gmax_kernel<<<blk(NB * 1024), T, 0, stream>>>(Y5, ss, gbuf);
    gcontrib_kernel<<<blk(NB * 512), T, 0, stream>>>(W[6], gbuf, C6);
    gemm((bf16*)(ws + o_W6f), xcat, Y6, 512, BN_, 192, 192, C6);
    stats(Y6, BN_, 512, G[6], Bb[6]);
    bn_apply_bf16k<<<blk((size_t)BN_ * 512), T, 0, stream>>>(Y6, ss, X6, BN_, 512);
    gemm((bf16*)(ws + o_W7), X6, Y7, 128, BN_, 512, 512, nullptr);
    stats(Y7, BN_, 128, G[7], Bb[7]);

    // ---- voxelize (scatter mean) ----
    hipMemsetAsync(acc, 0, (size_t)NB * R3_ * 128 * 4, stream);
    hipMemsetAsync(cnt, 0, (size_t)NB * R3_ * 4, stream);
    voxel_scatter<<<BN_, 128, 0, stream>>>(pts, Y7, ss, acc, cnt);
    voxel_final<<<blk((size_t)NB * 128 * R3_), T, 0, stream>>>(acc, cnt, out);
}
